// get_model_2095944040759
// MI455X (gfx1250) — matrix operand order repaired and checked by reading, not compiled, not run on hardware
//
#include <hip/hip_runtime.h>
#include <math.h>

#define BATCH 16
#define NPTS  4096

typedef __attribute__((ext_vector_type(16))) _Float16 v16h;
typedef __attribute__((ext_vector_type(8)))  _Float16 v8h;
typedef __attribute__((ext_vector_type(4)))  _Float16 v4h;
typedef __attribute__((ext_vector_type(8)))  float    v8f;
typedef __attribute__((ext_vector_type(4)))  float    v4f;

// ---------------------------------------------------------------------------
// init: xyz [B,9,N] -> l0_xyz [B,N,3], l0_pts [B,N,9]
// ---------------------------------------------------------------------------
__global__ void k_init_l0(const float* __restrict__ xin, float* __restrict__ l0xyz,
                          float* __restrict__ l0pts) {
  int gid = blockIdx.x * blockDim.x + threadIdx.x;
  if (gid >= BATCH * NPTS) return;
  int b = gid / NPTS, n = gid % NPTS;
#pragma unroll
  for (int c = 0; c < 9; c++) {
    float v = xin[((size_t)b * 9 + c) * NPTS + n];
    l0pts[(size_t)gid * 9 + c] = v;
    if (c < 3) l0xyz[(size_t)gid * 3 + c] = v;
  }
}

// ---------------------------------------------------------------------------
// Farthest point sampling: one block per batch, deterministic argmax
// (first-max tie-break to match jnp.argmax). dist persists in d_ws.
// ---------------------------------------------------------------------------
__global__ void k_fps(const float* __restrict__ xyz, int N, int S,
                      float* __restrict__ dist, int* __restrict__ out) {
  int b = blockIdx.x;
  const float* px = xyz + (size_t)b * N * 3;
  float* pd = dist + (size_t)b * N;
  __shared__ float sval[256];
  __shared__ int   sidx[256];
  int t = threadIdx.x;
  for (int i = t; i < N; i += 256) pd[i] = 1e10f;
  __syncthreads();
  int far = 0;
  for (int s = 0; s < S; s++) {
    if (t == 0) out[b * S + s] = far;
    float fx = px[far * 3 + 0], fy = px[far * 3 + 1], fz = px[far * 3 + 2];
    float best = -1.f; int bi = 0;
    for (int i = t; i < N; i += 256) {
      float dx = px[i * 3 + 0] - fx, dy = px[i * 3 + 1] - fy, dz = px[i * 3 + 2] - fz;
      float d = dx * dx + dy * dy + dz * dz;
      float nd = fminf(pd[i], d);
      pd[i] = nd;
      if (nd > best) { best = nd; bi = i; }   // strict > keeps lowest index
    }
    sval[t] = best; sidx[t] = bi;
    __syncthreads();
    for (int off = 128; off > 0; off >>= 1) {
      if (t < off) {
        if (sval[t + off] > sval[t] ||
            (sval[t + off] == sval[t] && sidx[t + off] < sidx[t])) {
          sval[t] = sval[t + off]; sidx[t] = sidx[t + off];
        }
      }
      __syncthreads();
    }
    far = sidx[0];
    __syncthreads();
  }
}

__global__ void k_gather(const float* __restrict__ xyz, const int* __restrict__ idx,
                         int N, int S, float* __restrict__ out) {
  int gid = blockIdx.x * blockDim.x + threadIdx.x;
  if (gid >= BATCH * S) return;
  int b = gid / S;
  int i = idx[gid];
  const float* p = xyz + ((size_t)b * N + i) * 3;
  out[(size_t)gid * 3 + 0] = p[0];
  out[(size_t)gid * 3 + 1] = p[1];
  out[(size_t)gid * 3 + 2] = p[2];
}

// ---------------------------------------------------------------------------
// Ball query: first nsample hits in ascending index order, pad with first hit
// ---------------------------------------------------------------------------
__global__ void k_ball(const float* __restrict__ xyz, const float* __restrict__ nxyz,
                       int N, int S, int ns, float r2, int* __restrict__ idx) {
  int gid = blockIdx.x * blockDim.x + threadIdx.x;
  if (gid >= BATCH * S) return;
  int b = gid / S;
  const float* q = nxyz + (size_t)gid * 3;
  float qx = q[0], qy = q[1], qz = q[2];
  const float* px = xyz + (size_t)b * N * 3;
  int* o = idx + (size_t)gid * ns;
  int cnt = 0, first = 0;
  for (int i = 0; i < N; i++) {
    float dx = px[i * 3 + 0] - qx, dy = px[i * 3 + 1] - qy, dz = px[i * 3 + 2] - qz;
    float d = dx * dx + dy * dy + dz * dz;
    if (d <= r2) {
      if (cnt == 0) first = i;
      o[cnt++] = i;
      if (cnt == ns) break;
    }
  }
  for (; cnt < ns; cnt++) o[cnt] = first;
}

// grouped features, K-padded: out row stride ldg (= round_up(3+D,32)),
// cols [3+D, ldg) zero-filled so the GEMM never needs a K guard on A.
__global__ void k_group(const float* __restrict__ xyz, const float* __restrict__ pts,
                        const float* __restrict__ nxyz, const int* __restrict__ idx,
                        int N, int S, int ns, int D, int ldg, float* __restrict__ out) {
  int gid = blockIdx.x * blockDim.x + threadIdx.x;
  int tot = BATCH * S * ns;
  if (gid >= tot) return;
  int rs = gid / ns;
  int b = rs / S;
  int pi = idx[gid];
  const float* p = xyz + ((size_t)b * N + pi) * 3;
  const float* q = nxyz + (size_t)rs * 3;
  float* o = out + (size_t)gid * ldg;
  o[0] = p[0] - q[0]; o[1] = p[1] - q[1]; o[2] = p[2] - q[2];
  const float* f = pts + ((size_t)b * N + pi) * D;
  for (int c = 0; c < D; c++) o[3 + c] = f[c];
  for (int c = 3 + D; c < ldg; c++) o[c] = 0.f;
}

// ---------------------------------------------------------------------------
// GEMM  Dst[M,N] = A[M,K] @ W[K,N] + bias   via v_wmma_f32_16x16x32_f16.
// Block = 128 threads = 4 waves; tile 64(M) x NT*16(N); K stepped by 32 over
// Kp = round_up(K,32) <= lda (A zero-padded to lda by the producer). Staging
// is split load-phase -> store-phase with distinct locals so the 4 A
// global_load_b128 (and 4*NT B loads) issue as a clause and retire with
// staged s_wait_loadcnt instead of serializing on one register quad.
// Each wave holds one A fragment and issues NT WMMAs against NT B fragments
// (A reuse cuts global A traffic NT-fold). Fragments are contiguous 8-half
// runs -> 16B LDS loads. EXEC uniform around the WMMAs (wave32, ISA 7.12).
// ---------------------------------------------------------------------------
template <int NT>
__global__ __launch_bounds__(128) void k_gemm(const float* __restrict__ A, int lda,
                                              const float* __restrict__ W,
                                              const float* __restrict__ bias,
                                              float* __restrict__ Dst,
                                              int M, int K, int N) {
  __shared__ _Float16 sA[64][32];        // [row][k]  row stride 64 B
  __shared__ _Float16 sBt[NT * 16][32];  // [col][k]  row stride 64 B
  int tid = threadIdx.x;
  int wave = tid >> 5, lane = tid & 31;
  int m0 = blockIdx.x * 64, n0 = blockIdx.y * (NT * 16);
  if (m0 >= M) return;

  v8f acc[NT] = {};
  int ar = (wave << 4) + (lane & 15);   // A row in tile for this lane
  int kb = (lane >> 4) << 3;            // A: lanes 16-31 hold K+8 halves
  int bn = lane & 15;                   // B column within a 16-wide subtile
  int bk = (lane >> 4) << 3;            // B: lanes 16-31 hold K+8 halves
  int Kp = (K + 31) & ~31;
  bool fullTile = (n0 + NT * 16 <= N);

  for (int k0 = 0; k0 < Kp; k0 += 32) {
    // ---- stage A: 64x32 floats; load phase into distinct locals first ----
    v4f va[4];
#pragma unroll
    for (int it = 0; it < 4; it++) {
      int chunk = tid + (it << 7);            // 0..511
      int r = chunk >> 3, c4 = (chunk & 7) << 2;
      va[it] = *(const v4f*)&A[(size_t)(m0 + r) * lda + k0 + c4];
    }
    // ---- stage B (transposed: [col][k]); load phase into locals ----
    float vb[NT * 4];
    if (fullTile && (k0 + 32 <= K)) {
#pragma unroll
      for (int it = 0; it < NT * 4; it++) {
        int i = tid + (it << 7);
        int r = i / (NT * 16), c = i % (NT * 16);
        vb[it] = W[(size_t)(k0 + r) * N + n0 + c];
      }
    } else {                                  // N=13 head, or last K-step when K%32 != 0
#pragma unroll
      for (int it = 0; it < NT * 4; it++) {
        int i = tid + (it << 7);
        int r = i / (NT * 16), c = i % (NT * 16);
        int kk = k0 + r, nn = n0 + c;
        vb[it] = (kk < K && nn < N) ? W[(size_t)kk * N + nn] : 0.f;
      }
    }
    // ---- convert + LDS store phase ----
#pragma unroll
    for (int it = 0; it < 4; it++) {
      int chunk = tid + (it << 7);
      int r = chunk >> 3, c4 = (chunk & 7) << 2;
      *(v4h*)&sA[r][c4] = __builtin_convertvector(va[it], v4h);
    }
#pragma unroll
    for (int it = 0; it < NT * 4; it++) {
      int i = tid + (it << 7);
      int r = i / (NT * 16), c = i % (NT * 16);
      sBt[c][r] = (_Float16)vb[it];
    }
    if (k0 + 32 < Kp)   // pull next A K-tile (global_prefetch_b8)
      __builtin_prefetch(&A[(size_t)(m0 + (tid & 63)) * lda + k0 + 32], 0, 3);
    __syncthreads();

    // ---- fragments: 16-byte LDS loads, stitched to v16h ----
    v8h al = *(const v8h*)&sA[ar][kb];
    v8h ah = *(const v8h*)&sA[ar][kb + 16];
    v16h a = __builtin_shufflevector(al, ah, 0, 1, 2, 3, 4, 5, 6, 7,
                                     8, 9, 10, 11, 12, 13, 14, 15);
#pragma unroll
    for (int nt = 0; nt < NT; nt++) {
      v8h bl = *(const v8h*)&sBt[nt * 16 + bn][bk];
      v8h bh = *(const v8h*)&sBt[nt * 16 + bn][bk + 16];
      v16h b = __builtin_shufflevector(bl, bh, 0, 1, 2, 3, 4, 5, 6, 7,
                                       8, 9, 10, 11, 12, 13, 14, 15);
      acc[nt] = __builtin_amdgcn_wmma_f32_16x16x32_f16(false, a, false, b, (short)0,
                                                       acc[nt], false, false);
    }
    __syncthreads();
  }

  int rbase = m0 + (wave << 4) + ((lane >> 4) << 3);  // D: row = 8*(lane>=16)+v
#pragma unroll
  for (int nt = 0; nt < NT; nt++) {
    int col = n0 + nt * 16 + bn;
    if (col < N) {
      float bv = bias[col];
#pragma unroll
      for (int v = 0; v < 8; v++)
        Dst[(size_t)(rbase + v) * N + col] = acc[nt][v] + bv;
    }
  }
}

// ---------------------------------------------------------------------------
// BN stats: one block per channel, fixed-association tree sum (deterministic).
// st[c] = sum, st[N+c] = sum of squares.
// ---------------------------------------------------------------------------
__global__ void k_bn_stats(const float* __restrict__ x, int M, int N,
                           float* __restrict__ st) {
  int c = blockIdx.x;
  int t = threadIdx.x;
  float s = 0.f, s2 = 0.f;
  for (int r = t; r < M; r += 256) {
    float v = x[(size_t)r * N + c];
    s += v; s2 += v * v;
  }
  __shared__ float sa[256], sb[256];
  sa[t] = s; sb[t] = s2;
  __syncthreads();
  for (int off = 128; off > 0; off >>= 1) {
    if (t < off) { sa[t] += sa[t + off]; sb[t] += sb[t + off]; }
    __syncthreads();
  }
  if (t == 0) { st[c] = sa[0]; st[N + c] = sb[0]; }
}

__global__ void k_bn_relu(float* __restrict__ x, int M, int N,
                          const float* __restrict__ st, const float* __restrict__ g,
                          const float* __restrict__ be) {
  long gid = (long)blockIdx.x * blockDim.x + threadIdx.x;
  if (gid >= (long)M * N) return;
  int c = (int)(gid % N);
  float inv = 1.0f / (float)M;
  float mean = st[c] * inv;
  float var = st[N + c] * inv - mean * mean;
  float sc = g[c] * rsqrtf(var + 1e-5f);
  float v = (x[gid] - mean) * sc + be[c];
  x[gid] = v > 0.f ? v : 0.f;
}

// max over neighborhood: in [B*S*ns, C] -> out [B*S, C]
__global__ void k_maxpool(const float* __restrict__ in, int S, int ns, int C,
                          float* __restrict__ out) {
  long gid = (long)blockIdx.x * blockDim.x + threadIdx.x;
  long tot = (long)BATCH * S * C;
  if (gid >= tot) return;
  int c = (int)(gid % C);
  long rs = gid / C;
  const float* p = in + rs * (long)ns * C + c;
  float m = -3.4e38f;
  for (int j = 0; j < ns; j++) m = fmaxf(m, p[(long)j * C]);
  out[gid] = m;
}

// ---------------------------------------------------------------------------
// FP: 3-NN inverse-distance interpolation (stable tie-break = lowest index)
// ---------------------------------------------------------------------------
__global__ void k_interp3(const float* __restrict__ xyz1, const float* __restrict__ xyz2,
                          const float* __restrict__ p2, int N, int S, int D,
                          float* __restrict__ out) {
  int gid = blockIdx.x * blockDim.x + threadIdx.x;
  if (gid >= BATCH * N) return;
  int b = gid / N;
  const float* q = xyz1 + (size_t)gid * 3;
  float qx = q[0], qy = q[1], qz = q[2];
  const float* base = xyz2 + (size_t)b * S * 3;
  float d0 = 3.4e38f, d1 = 3.4e38f, d2 = 3.4e38f;
  int i0 = 0, i1 = 0, i2 = 0;
  for (int s = 0; s < S; s++) {
    float dx = base[s * 3 + 0] - qx, dy = base[s * 3 + 1] - qy, dz = base[s * 3 + 2] - qz;
    float d = dx * dx + dy * dy + dz * dz;
    if (d < d0)      { d2 = d1; i2 = i1; d1 = d0; i1 = i0; d0 = d; i0 = s; }
    else if (d < d1) { d2 = d1; i2 = i1; d1 = d;  i1 = s; }
    else if (d < d2) { d2 = d;  i2 = s; }
  }
  float w0 = 1.f / (d0 + 1e-8f), w1 = 1.f / (d1 + 1e-8f), w2 = 1.f / (d2 + 1e-8f);
  float wsum = w0 + w1 + w2;
  w0 /= wsum; w1 /= wsum; w2 /= wsum;
  const float* r0 = p2 + ((size_t)b * S + i0) * D;
  const float* r1 = p2 + ((size_t)b * S + i1) * D;
  const float* r2 = p2 + ((size_t)b * S + i2) * D;
  float* o = out + (size_t)gid * D;
  for (int c = 0; c < D; c++) o[c] = w0 * r0[c] + w1 * r1[c] + w2 * r2[c];
}

// concat rows: out[R, Da+Db] = [a | b]; Da may be 0 (a never read then)
__global__ void k_concat(const float* __restrict__ a, const float* __restrict__ bsrc,
                         int R, int Da, int Db, float* __restrict__ out) {
  long gid = (long)blockIdx.x * blockDim.x + threadIdx.x;
  int Dc = Da + Db;
  if (gid >= (long)R * Dc) return;
  int c = (int)(gid % Dc);
  long r = gid / Dc;
  out[gid] = (c < Da) ? a[r * Da + c] : bsrc[r * Db + (c - Da)];
}

__global__ void k_logsm(const float* __restrict__ in, int R, float* __restrict__ out) {
  int r = blockIdx.x * blockDim.x + threadIdx.x;
  if (r >= R) return;
  const float* x = in + (size_t)r * 13;
  float m = x[0];
#pragma unroll
  for (int c = 1; c < 13; c++) m = fmaxf(m, x[c]);
  float ssum = 0.f;
#pragma unroll
  for (int c = 0; c < 13; c++) ssum += expf(x[c] - m);
  float l = logf(ssum);
#pragma unroll
  for (int c = 0; c < 13; c++) out[(size_t)r * 13 + c] = x[c] - m - l;
}

// p [B,N,128] -> L2-normalized, transposed to [B,128,N]
__global__ void k_norm_feat(const float* __restrict__ p, int N, float* __restrict__ out) {
  int gid = blockIdx.x * blockDim.x + threadIdx.x;
  if (gid >= BATCH * N) return;
  int b = gid / N, n = gid % N;
  const float* row = p + (size_t)gid * 128;
  float s = 0.f;
  for (int c = 0; c < 128; c++) s += row[c] * row[c];
  float inv = 1.f / fmaxf(sqrtf(s), 1e-12f);
  for (int c = 0; c < 128; c++)
    out[(size_t)b * 128 * N + (size_t)c * N + n] = row[c] * inv;
}

// ---------------------------------------------------------------------------
// host orchestration
// ---------------------------------------------------------------------------
struct Lyr { const float *W, *b, *g, *be; int ci, co; };

extern "C" void kernel_launch(void* const* d_in, const int* in_sizes, int n_in,
                              void* d_out, int out_size, void* d_ws, size_t ws_size,
                              hipStream_t stream) {
  (void)in_sizes; (void)n_in; (void)out_size; (void)ws_size;

  // ---- parse inputs (insertion-order recursive flatten of setup_inputs) ----
  auto F = [&](int i) { return (const float*)d_in[i]; };
  int ti = 1;  // 0 = xyz
  static const int sadim[4][4] = {{12,32,32,64},{67,64,64,128},
                                  {131,128,128,256},{259,256,256,512}};
  Lyr saL[4][3];
  for (int l = 0; l < 4; l++)
    for (int j = 0; j < 3; j++) {
      saL[l][j] = {F(ti), F(ti+1), F(ti+2), F(ti+3), sadim[l][j], sadim[l][j+1]};
      ti += 4;
    }
  static const int fpdim[4][4] = {{768,256,256,256},{384,256,256,256},
                                  {320,256,128,128},{128,128,128,128}};
  static const int fpn[4] = {2,2,2,3};
  Lyr fpL[4][3];
  for (int l = 0; l < 4; l++)
    for (int j = 0; j < fpn[l]; j++) {
      fpL[l][j] = {F(ti), F(ti+1), F(ti+2), F(ti+3), fpdim[l][j], fpdim[l][j+1]};
      ti += 4;
    }
  Lyr h1 = {F(ti), F(ti+1), F(ti+2), F(ti+3), 128, 128}; ti += 4;
  const float* h2W = F(ti); const float* h2b = F(ti+1);
  const float* xin = (const float*)d_in[0];

  // ---- workspace carve ----
  size_t woff = 0;
  auto alloc = [&](size_t bytes) -> void* {
    woff = (woff + 255) & ~(size_t)255;
    void* p = (char*)d_ws + woff;
    woff += bytes;
    return p;
  };
  auto af = [&](size_t n) { return (float*)alloc(n * sizeof(float)); };

  float* l0xyz = af((size_t)BATCH * NPTS * 3);
  float* l0pts = af((size_t)BATCH * NPTS * 9);
  float* xyzs[4]  = {af(16*1024*3), af(16*256*3), af(16*64*3), af(16*16*3)};
  float* feats[4] = {af(16*1024*64), af(16*256*128), af(16*64*256), af(16*16*512)};
  float* fpo[4]   = {af(1024*256), af(4096*256), af(16384*128), af((size_t)65536*128)};
  float* fpsdist  = af((size_t)BATCH * NPTS);
  int*   fpsidx   = (int*)alloc((size_t)BATCH * 1024 * sizeof(int));
  int*   ballidx  = (int*)alloc((size_t)BATCH * 1024 * 32 * sizeof(int));
  float* interp   = af((size_t)BATCH * NPTS * 128);
  float* stats    = af(1024);
  float* bufA     = af((size_t)524288 * 64);
  float* bufB     = af((size_t)524288 * 64);

  // ---- GEMM dispatch: widest N-tile the layer supports ----
  auto launch_gemm = [&](const float* in, int lda, const float* W, const float* b,
                         float* out, int M, int K, int N) {
    int nt = (N >= 64) ? 4 : (N >= 32 ? 2 : 1);
    dim3 g(M / 64, (N + nt * 16 - 1) / (nt * 16));
    switch (nt) {
      case 4: k_gemm<4><<<g, 128, 0, stream>>>(in, lda, W, b, out, M, K, N); break;
      case 2: k_gemm<2><<<g, 128, 0, stream>>>(in, lda, W, b, out, M, K, N); break;
      default: k_gemm<1><<<g, 128, 0, stream>>>(in, lda, W, b, out, M, K, N); break;
    }
  };

  // ---- layer runner: WMMA GEMM + deterministic BN + ReLU ----
  auto run_layer = [&](const float* in, int lda, float* out, int M, const Lyr& L) {
    launch_gemm(in, lda, L.W, L.b, out, M, L.ci, L.co);
    k_bn_stats<<<L.co, 256, 0, stream>>>(out, M, L.co, stats);
    long tot = (long)M * L.co;
    k_bn_relu<<<(int)((tot + 255) / 256), 256, 0, stream>>>(out, M, L.co, stats,
                                                            L.g, L.be);
  };

  // ---- set abstraction ----
  auto run_sa = [&](const float* xyz, const float* pts, int N, int D, int S,
                    float radius, int ns, Lyr* mlp, float* oxyz, float* ofeat) {
    k_fps<<<BATCH, 256, 0, stream>>>(xyz, N, S, fpsdist, fpsidx);
    int bs = BATCH * S;
    k_gather<<<(bs + 255) / 256, 256, 0, stream>>>(xyz, fpsidx, N, S, oxyz);
    k_ball<<<(bs + 255) / 256, 256, 0, stream>>>(xyz, oxyz, N, S, ns,
                                                 radius * radius, ballidx);
    int rows = bs * ns;
    int ldg = (3 + D + 31) & ~31;          // K-padded grouped stride
    k_group<<<(rows + 255) / 256, 256, 0, stream>>>(xyz, pts, oxyz, ballidx,
                                                    N, S, ns, D, ldg, bufA);
    float* cur = bufA; float* nxt = bufB;
    int lda = ldg;
    for (int j = 0; j < 3; j++) {
      run_layer(cur, lda, nxt, rows, mlp[j]);
      lda = mlp[j].co;                     // subsequent K's are %32 == 0
      float* t2 = cur; cur = nxt; nxt = t2;
    }
    int co = mlp[2].co;
    long tot = (long)bs * co;
    k_maxpool<<<(int)((tot + 255) / 256), 256, 0, stream>>>(cur, S, ns, co, ofeat);
  };

  // ---- feature propagation ----
  auto run_fp = [&](const float* xyz1, int N1, const float* xyz2, int S2,
                    const float* pts1, int D1, const float* pts2, int D2,
                    Lyr* mlp, int nl, float* outp) {
    int rows = BATCH * N1;
    k_interp3<<<(rows + 255) / 256, 256, 0, stream>>>(xyz1, xyz2, pts2,
                                                      N1, S2, D2, interp);
    long tot = (long)rows * (D1 + D2);
    k_concat<<<(int)((tot + 255) / 256), 256, 0, stream>>>(pts1, interp, rows,
                                                           D1, D2, bufA);
    float* cur = bufA; float* nxt = bufB;
    int lda = D1 + D2;                     // all %32 == 0
    for (int j = 0; j < nl; j++) {
      run_layer(cur, lda, nxt, rows, mlp[j]);
      lda = mlp[j].co;
      float* t2 = cur; cur = nxt; nxt = t2;
    }
    hipMemcpyAsync(outp, cur, (size_t)rows * mlp[nl - 1].co * sizeof(float),
                   hipMemcpyDeviceToDevice, stream);
  };

  // ---- pipeline ----
  k_init_l0<<<(BATCH * NPTS + 255) / 256, 256, 0, stream>>>(xin, l0xyz, l0pts);

  struct { int S; float r; } sac[4] = {{1024, 0.1f}, {256, 0.2f}, {64, 0.4f}, {16, 0.8f}};
  const float* cxyz = l0xyz; const float* cpts = l0pts;
  int curN = NPTS, curD = 9;
  for (int l = 0; l < 4; l++) {
    run_sa(cxyz, cpts, curN, curD, sac[l].S, sac[l].r, 32, saL[l], xyzs[l], feats[l]);
    cxyz = xyzs[l]; cpts = feats[l]; curN = sac[l].S; curD = sadim[l][3];
  }

  run_fp(xyzs[2],   64, xyzs[3],   16, feats[2], 256, feats[3], 512, fpL[0], 2, fpo[0]);
  run_fp(xyzs[1],  256, xyzs[2],   64, feats[1], 128, fpo[0],   256, fpL[1], 2, fpo[1]);
  run_fp(xyzs[0], 1024, xyzs[1],  256, feats[0],  64, fpo[1],   256, fpL[2], 2, fpo[2]);
  run_fp(l0xyz,   4096, xyzs[0], 1024, nullptr,    0, fpo[2],   128, fpL[3], 3, fpo[3]);

  // ---- head ----
  int R = BATCH * NPTS;
  run_layer(fpo[3], 128, bufA, R, h1);
  launch_gemm(bufA, 128, h2W, h2b, bufB, R, 128, 13);
  float* out_f = (float*)d_out;
  k_logsm<<<(R + 255) / 256, 256, 0, stream>>>(bufB, R, out_f);
  k_norm_feat<<<(R + 255) / 256, 256, 0, stream>>>(fpo[3], NPTS,
                                                   out_f + (size_t)R * 13);
}
